// ImprovedHANModel_32899449488091
// MI455X (gfx1250) — matrix operand order repaired and checked by reading, not compiled, not run on hardware
//
#include <hip/hip_runtime.h>
#include <hip/hip_bf16.h>

typedef __attribute__((ext_vector_type(16))) _Float16 v16h;
typedef __attribute__((ext_vector_type(8)))  float    v8f;

#define NP_N 100000
#define NA_N 50000
#define NK_N 10000
#define IN_D 128
#define HID_D 32
#define H_N 8
#define HD_D 256
#define OUT_D 64
#define NB 4   // N-tiles (16 cols each) per wave -> 16x64 wave tile

// ---------- helpers ----------
static __device__ __forceinline__ float lrelu(float x) { return x > 0.0f ? x : 0.2f * x; }

// monotone order-preserving float<->uint encoding: segment-max via one HW uint atomicMax
static __device__ __forceinline__ unsigned encf(float f) {
  unsigned u = __float_as_uint(f);
  return (u & 0x80000000u) ? ~u : (u | 0x80000000u);
}
static __device__ __forceinline__ float decf(unsigned u) {
  return (u & 0x80000000u) ? __uint_as_float(u & 0x7FFFFFFFu) : __uint_as_float(~u);
}

// ---------- elementwise / prep kernels ----------
__global__ void k_cvt_f16(const float* __restrict__ x, _Float16* __restrict__ y, long n) {
  long i = (long)blockIdx.x * blockDim.x + threadIdx.x;
  if (i < n) y[i] = (_Float16)x[i];
}

__global__ void k_fill_u32(unsigned* __restrict__ p, unsigned v, long n) {
  long i = (long)blockIdx.x * blockDim.x + threadIdx.x;
  if (i < n) p[i] = v;
}

// W [K x N] f32 row-major -> Wt [N x K] f16 row-major
__global__ void k_transpose_f16(const float* __restrict__ W, _Float16* __restrict__ Wt,
                                int K, int N) {
  long t = (long)blockIdx.x * blockDim.x + threadIdx.x;
  if (t >= (long)K * N) return;
  int n = (int)(t / K);
  int k = (int)(t - (long)n * K);
  Wt[t] = (_Float16)W[(long)k * N + n];
}

// wa[c*8+h] = sum_d W[c, h*32+d] * a[h*32+d]   (folds attention vector through W)
__global__ void k_fold_attn(const float* __restrict__ W, const float* __restrict__ a,
                            float* __restrict__ wa, int K) {
  int t = blockIdx.x * blockDim.x + threadIdx.x;
  if (t >= K * H_N) return;
  int c = t >> 3, h = t & 7;
  float s = 0.0f;
#pragma unroll
  for (int d = 0; d < HID_D; ++d) s += W[(long)c * HD_D + h * HID_D + d] * a[h * HID_D + d];
  wa[t] = s;
}

// e[n*8+h] = sum_c x[n,c] * wa[c*8+h]
__global__ void k_attn_proj(const _Float16* __restrict__ x, const float* __restrict__ wa,
                            float* __restrict__ e, int N, int K) {
  long t = (long)blockIdx.x * blockDim.x + threadIdx.x;
  if (t >= (long)N * H_N) return;
  int n = (int)(t >> 3), h = (int)(t & 7);
  const _Float16* xr = x + (long)n * K;
  float s = 0.0f;
  for (int c = 0; c < K; ++c) s += (float)xr[c] * wa[c * H_N + h];
  e[t] = s;
}

// ---------- WMMA GEMM: C[MxN] = A[MxK](f16) @ Bt[NxK](f16)^T (+bias) ----------
// Wave tile = 16(M) x 64(N): 4 accumulators share one A fragment; K stepped by 64
// so each iteration issues one load batch then 8 independent back-to-back WMMAs.
// Requires N % 64 == 0, M % 16 == 0, K % 64 == 0 (true for all call sites).
__global__ void k_gemm_wmma(const _Float16* __restrict__ A, const _Float16* __restrict__ Bt,
                            const float* __restrict__ bias, float* __restrict__ C,
                            int M, int N, int K) {
  const int lane = threadIdx.x & 31;
  const int wave = threadIdx.x >> 5;
  const int groupsN = N >> 6;
  const long totalW = (long)(M >> 4) * groupsN;
  long wt = (long)blockIdx.x * 8 + wave;
  if (wt >= totalW) return;                 // wave-uniform; EXEC stays all-ones
  const int tm = (int)(wt / groupsN);
  const int gn = (int)(wt % groupsN);

  const int rowA = (tm << 4) + (lane & 15); // ISA 16-bit A layout: lane -> M row
  const unsigned* __restrict__ Ap = (const unsigned*)(A + (size_t)rowA * K);
  const unsigned* __restrict__ Bp[NB];
#pragma unroll
  for (int j = 0; j < NB; ++j) {
    int rowB = (gn << 6) + (j << 4) + (lane & 15);   // Bt row = output column
    Bp[j] = (const unsigned*)(Bt + (size_t)rowB * K);
  }
  const int hiA = (lane & 16) ? 8 : 0;      // lanes16-31: K offset +8 (A layout)
  const int hiB = (lane & 16) ? 8 : 0;      // lanes16-31: K offset +8 (B layout)

  v8f acc[NB] = {};
  union F { v16h h; unsigned u[8]; };

  for (int kk = 0; kk < K; kk += 64) {
    F fa0, fa1, fb0[NB], fb1[NB];
#pragma unroll
    for (int v = 0; v < 8; ++v) {
      int ka = ((v & 4) ? 16 : 0) + hiA + ((v & 3) << 1);
      fa0.u[v] = Ap[(kk + ka) >> 1];
      fa1.u[v] = Ap[(kk + 32 + ka) >> 1];
      int kb = ((v & 4) ? 16 : 0) + hiB + ((v & 3) << 1);
#pragma unroll
      for (int j = 0; j < NB; ++j) {
        fb0[j].u[v] = Bp[j][(kk + kb) >> 1];
        fb1[j].u[v] = Bp[j][(kk + 32 + kb) >> 1];
      }
    }
    if (kk + 64 < K)                         // global_prefetch_b8 next A slab
      __builtin_prefetch((const void*)(Ap + ((kk + 64) >> 1)), 0, 0);
#pragma unroll
    for (int j = 0; j < NB; ++j)
      acc[j] = __builtin_amdgcn_wmma_f32_16x16x32_f16(false, fa0.h, false, fb0[j].h,
                                                      (short)0, acc[j], false, false);
#pragma unroll
    for (int j = 0; j < NB; ++j)
      acc[j] = __builtin_amdgcn_wmma_f32_16x16x32_f16(false, fa1.h, false, fb1[j].h,
                                                      (short)0, acc[j], false, false);
  }

  const int row0 = (tm << 4) + ((lane & 16) ? 8 : 0);  // C layout: VGPR i -> M=i / i+8
#pragma unroll
  for (int j = 0; j < NB; ++j) {
    const int col = (gn << 6) + (j << 4) + (lane & 15);
    const float badd = bias ? bias[col] : 0.0f;
#pragma unroll
    for (int i = 0; i < 8; ++i)
      C[(size_t)(row0 + i) * N + col] = acc[j][i] + badd;
  }
}

// ---------- edge softmax passes ----------
__global__ void k_edge_max(const int* __restrict__ src, const int* __restrict__ dst,
                           const float* __restrict__ el, const float* __restrict__ er,
                           unsigned* __restrict__ m, long EH) {
  long t = (long)blockIdx.x * blockDim.x + threadIdx.x;
  if (t >= EH) return;
  long e = t >> 3; int h = (int)(t & 7);
  int si = src[e], di = dst[e];
  float v = lrelu(el[(long)si * 8 + h] + er[(long)di * 8 + h]);
  atomicMax(&m[(long)di * 8 + h], encf(v));
}

__global__ void k_edge_sum(const int* __restrict__ src, const int* __restrict__ dst,
                           const float* __restrict__ el, const float* __restrict__ er,
                           const unsigned* __restrict__ m, float* __restrict__ s, long EH) {
  long t = (long)blockIdx.x * blockDim.x + threadIdx.x;
  if (t >= EH) return;
  long e = t >> 3; int h = (int)(t & 7);
  int si = src[e], di = dst[e];
  float v = lrelu(el[(long)si * 8 + h] + er[(long)di * 8 + h]);
  float ex = __expf(v - decf(m[(long)di * 8 + h]));
  atomicAdd(&s[(long)di * 8 + h], ex);
}

__global__ void k_edge_scatter(const int* __restrict__ src, const int* __restrict__ dst,
                               const float* __restrict__ el, const float* __restrict__ er,
                               const unsigned* __restrict__ m, const float* __restrict__ s,
                               const float* __restrict__ hs, float* __restrict__ out, long EH) {
  long t = (long)blockIdx.x * blockDim.x + threadIdx.x;
  if (t >= EH) return;
  long e = t >> 3; int h = (int)(t & 7);
  int si = src[e], di = dst[e];
  float v = lrelu(el[(long)si * 8 + h] + er[(long)di * 8 + h]);
  float alpha = __expf(v - decf(m[(long)di * 8 + h])) / fmaxf(s[(long)di * 8 + h], 1e-9f);
  const float* __restrict__ hsr = hs + (long)si * HD_D + h * HID_D;
  float* __restrict__ outr = out + (long)di * HD_D + h * HID_D;
#pragma unroll
  for (int d = 0; d < HID_D; ++d) atomicAdd(&outr[d], alpha * hsr[d]);
}

// add per-relation biases (up to 2 relations target this node type) + optional ELU
__global__ void k_bias_act(float* __restrict__ hbuf, const float* __restrict__ b1,
                           const float* __restrict__ b2, long n, int doElu) {
  long t = (long)blockIdx.x * blockDim.x + threadIdx.x;
  if (t >= n) return;
  int c = (int)(t & (HD_D - 1));
  float v = hbuf[t] + b1[c] + (b2 ? b2[c] : 0.0f);
  if (doElu) v = v > 0.0f ? v : expm1f(v);
  hbuf[t] = v;
}

// ---------- host orchestration ----------
extern "C" void kernel_launch(void* const* d_in, const int* in_sizes, int n_in,
                              void* d_out, int out_size, void* d_ws, size_t ws_size,
                              hipStream_t stream) {
  (void)in_sizes; (void)n_in; (void)out_size; (void)ws_size;
  const int cnt[3] = {NP_N, NA_N, NK_N};           // 0=p, 1=a, 2=k
  const float* xf[3] = {(const float*)d_in[0], (const float*)d_in[1], (const float*)d_in[2]};

  struct RelDef { int s, d; const int* src; const int* dst; long E; int base; };
  const RelDef rel[4] = {
    {1, 0, (const int*)d_in[3], (const int*)d_in[4], 400000, 11},  // ap: a->p
    {0, 1, (const int*)d_in[5], (const int*)d_in[6], 400000, 19},  // pa: p->a
    {0, 2, (const int*)d_in[7], (const int*)d_in[8], 200000, 27},  // pk: p->k
    {2, 0, (const int*)d_in[9], (const int*)d_in[10], 200000, 35}, // kp: k->p
  };
  // dict order per relation: W1(base) al1 ar1 b1 W2(base+4) al2 ar2 b2
  const float* Wo = (const float*)d_in[43];
  const float* bo = (const float*)d_in[44];

  char* w = (char*)d_ws; size_t off = 0;
  auto alloc = [&](size_t bytes) -> void* {
    void* p = w + off; off = (off + bytes + 255) & ~((size_t)255); return p;
  };
  _Float16* x16l1[3]; for (int t = 0; t < 3; ++t) x16l1[t] = (_Float16*)alloc((size_t)cnt[t] * IN_D * 2);
  _Float16* x16l2[3]; for (int t = 0; t < 3; ++t) x16l2[t] = (_Float16*)alloc((size_t)cnt[t] * HD_D * 2);
  _Float16* W1t[4];   for (int r = 0; r < 4; ++r) W1t[r] = (_Float16*)alloc((size_t)HD_D * IN_D * 2);
  _Float16* W2t[4];   for (int r = 0; r < 4; ++r) W2t[r] = (_Float16*)alloc((size_t)HD_D * HD_D * 2);
  _Float16* Wot = (_Float16*)alloc((size_t)OUT_D * HD_D * 2);
  float* wal = (float*)alloc((size_t)HD_D * H_N * 4);
  float* war = (float*)alloc((size_t)HD_D * H_N * 4);
  float* el  = (float*)alloc((size_t)NP_N * H_N * 4);
  float* er  = (float*)alloc((size_t)NP_N * H_N * 4);
  unsigned* mbuf = (unsigned*)alloc((size_t)NP_N * H_N * 4);
  float* sbuf = (float*)alloc((size_t)NP_N * H_N * 4);
  float* hs   = (float*)alloc((size_t)NP_N * HD_D * 4);
  float* hbuf[3]; for (int t = 0; t < 3; ++t) hbuf[t] = (float*)alloc((size_t)cnt[t] * HD_D * 4);

  auto nb = [](long n) { return (unsigned)((n + 255) / 256); };
  auto gemm_blocks = [](long M, long N) { return (unsigned)(((M / 16) * (N / 64) + 7) / 8); };

  // prep: f16 inputs + transposed f16 weights
  for (int t = 0; t < 3; ++t) {
    long n = (long)cnt[t] * IN_D;
    k_cvt_f16<<<nb(n), 256, 0, stream>>>(xf[t], x16l1[t], n);
  }
  for (int r = 0; r < 4; ++r) {
    k_transpose_f16<<<nb((long)HD_D * IN_D), 256, 0, stream>>>(
        (const float*)d_in[rel[r].base], W1t[r], IN_D, HD_D);
    k_transpose_f16<<<nb((long)HD_D * HD_D), 256, 0, stream>>>(
        (const float*)d_in[rel[r].base + 4], W2t[r], HD_D, HD_D);
  }
  k_transpose_f16<<<nb((long)OUT_D * HD_D), 256, 0, stream>>>(Wo, Wot, HD_D, OUT_D);

  for (int layer = 0; layer < 2; ++layer) {
    const int K = (layer == 0) ? IN_D : HD_D;
    for (int t = 0; t < 3; ++t) {
      long n = (long)cnt[t] * HD_D;
      k_fill_u32<<<nb(n), 256, 0, stream>>>((unsigned*)hbuf[t], 0u, n);
    }
    for (int r = 0; r < 4; ++r) {
      const RelDef& R = rel[r];
      const float* W  = (const float*)d_in[R.base + (layer ? 4 : 0)];
      const float* al = (const float*)d_in[R.base + (layer ? 5 : 1)];
      const float* ar = (const float*)d_in[R.base + (layer ? 6 : 2)];
      const _Float16* Wt   = layer ? W2t[r] : W1t[r];
      const _Float16* xs16 = layer ? x16l2[R.s] : x16l1[R.s];
      const _Float16* xd16 = layer ? x16l2[R.d] : x16l1[R.d];
      const int Ns = cnt[R.s], Nd = cnt[R.d];

      k_fold_attn<<<nb((long)K * H_N), 256, 0, stream>>>(W, al, wal, K);
      k_fold_attn<<<nb((long)K * H_N), 256, 0, stream>>>(W, ar, war, K);
      k_attn_proj<<<nb((long)Ns * H_N), 256, 0, stream>>>(xs16, wal, el, Ns, K);
      k_attn_proj<<<nb((long)Nd * H_N), 256, 0, stream>>>(xd16, war, er, Nd, K);

      k_gemm_wmma<<<gemm_blocks(Ns, HD_D), 256, 0, stream>>>(
          xs16, Wt, nullptr, hs, Ns, HD_D, K);

      k_fill_u32<<<nb((long)Nd * H_N), 256, 0, stream>>>(mbuf, 0x007FFFFFu, (long)Nd * H_N); // enc(-inf)
      k_fill_u32<<<nb((long)Nd * H_N), 256, 0, stream>>>((unsigned*)sbuf, 0u, (long)Nd * H_N);

      const long EH = R.E * H_N;
      k_edge_max<<<nb(EH), 256, 0, stream>>>(R.src, R.dst, el, er, mbuf, EH);
      k_edge_sum<<<nb(EH), 256, 0, stream>>>(R.src, R.dst, el, er, mbuf, sbuf, EH);
      k_edge_scatter<<<nb(EH), 256, 0, stream>>>(R.src, R.dst, el, er, mbuf, sbuf, hs, hbuf[R.d], EH);
    }
    const int bofs = layer ? 7 : 3;
    const float* bp1 = (const float*)d_in[rel[0].base + bofs]; // ap -> p
    const float* bp2 = (const float*)d_in[rel[3].base + bofs]; // kp -> p
    const float* ba  = (const float*)d_in[rel[1].base + bofs]; // pa -> a
    const float* bk  = (const float*)d_in[rel[2].base + bofs]; // pk -> k
    const int doElu = (layer == 0);
    k_bias_act<<<nb((long)NP_N * HD_D), 256, 0, stream>>>(hbuf[0], bp1, bp2, (long)NP_N * HD_D, doElu);
    k_bias_act<<<nb((long)NA_N * HD_D), 256, 0, stream>>>(hbuf[1], ba, nullptr, (long)NA_N * HD_D, doElu);
    k_bias_act<<<nb((long)NK_N * HD_D), 256, 0, stream>>>(hbuf[2], bk, nullptr, (long)NK_N * HD_D, doElu);
    for (int t = 0; t < 3; ++t) {
      long n = (long)cnt[t] * HD_D;
      k_cvt_f16<<<nb(n), 256, 0, stream>>>(hbuf[t], x16l2[t], n);
    }
  }

  // final projection: [N x 256] @ [256 x 64] + bo, written straight into d_out segments
  float* outp = (float*)d_out;
  long o = 0;
  for (int t = 0; t < 3; ++t) {
    k_gemm_wmma<<<gemm_blocks(cnt[t], OUT_D), 256, 0, stream>>>(
        x16l2[t], Wot, bo, outp + o, cnt[t], OUT_D, HD_D);
    o += (long)cnt[t] * OUT_D;
  }
}
